// CausalSelfAttention_61512521613733
// MI455X (gfx1250) — matrix operand order repaired and checked by reading, not compiled, not run on hardware
//
#include <hip/hip_runtime.h>

typedef __bf16 bf16;
typedef __attribute__((ext_vector_type(16))) __bf16 v16bf;
typedef __attribute__((ext_vector_type(8)))  __bf16 v8bf;
typedef __attribute__((ext_vector_type(8)))  float  v8f;
typedef __attribute__((ext_vector_type(4)))  unsigned int v4u;
typedef __attribute__((ext_vector_type(8)))  int v8i;
typedef __attribute__((ext_vector_type(4)))  int v4i;

constexpr int L_SEQ  = 2048;
constexpr int DIM    = 2048;
constexpr int NHEADS = 16;
constexpr int HDIM   = 128;

constexpr int BM = 128;
constexpr int BN = 64;
constexpr int BK = 32;

// ---------------------------------------------------------------------------
// WMMA helpers (gfx1250: V_WMMA_F32_16X16X32_BF16, wave32)
// ---------------------------------------------------------------------------
__device__ inline v8f wmma_bf16(v16bf a, v16bf b, v8f c) {
  return __builtin_amdgcn_wmma_f32_16x16x32_bf16(
      /*neg_a=*/false, a, /*neg_b=*/false, b,
      /*c_mod=*/(short)0, c, /*reuse_a=*/false, /*reuse_b=*/false);
}

// A fragment (16x32 bf16): lane = h*16+lm holds row (row0+lm);
// elements 0..7  = K[k0 + 8h .. +7], elements 8..15 = K[k0 + 16 + 8h .. +7]
__device__ inline v16bf frag_a_ld(const bf16* base, int ld, int row0, int k0,
                                  int lm, int h) {
  const bf16* p = base + (row0 + lm) * ld + k0 + 8 * h;
  union { v16bf v; v8bf u[2]; } r;
  r.u[0] = *(const v8bf*)(p);
  r.u[1] = *(const v8bf*)(p + 16);
  return r.v;
}

// B fragment (32x16 bf16): lane = h*16+lm holds column (col0+lm);
// elements 0..7  = K[k0 + 8h .. +7], elements 8..15 = K[k0 + 16 + 8h .. +7]
__device__ inline v16bf frag_b_ld(const bf16* base, int ld, int col0, int k0,
                                  int lm, int h) {
  const bf16* p = base + (col0 + lm) * ld + k0 + 8 * h;
  union { v16bf v; v8bf u[2]; } r;
  r.u[0] = *(const v8bf*)(p);
  r.u[1] = *(const v8bf*)(p + 16);
  return r.v;
}

// ---------------------------------------------------------------------------
// Tensor Data Mover: 2D tile (tile_w x tile_h bf16 elements, row stride
// stride_elems) global -> LDS, packed contiguously row-major in LDS.
// D# layout per CDNA5 ISA ch.8 (group0: count/lds/global/type; group1:
// data_size, tensor dims, tile dims, dim0 stride). Issue from ONE wave only
// (TDM ignores EXEC, issues per-wave); completion via s_wait_tensorcnt.
// ---------------------------------------------------------------------------
__device__ inline void tdm_load_2d(const void* gptr, const void* lptr,
                                   unsigned tile_w, unsigned tile_h,
                                   unsigned stride_elems,
                                   unsigned tensor_w, unsigned tensor_h) {
  unsigned long long ga = (unsigned long long)gptr;
  unsigned la = (unsigned)(unsigned long long)lptr;  // low 32 bits = LDS offset
  v4u g0;
  g0[0] = 1u;                                            // count=1, no gather
  g0[1] = la;                                            // lds_addr
  g0[2] = (unsigned)ga;                                  // global_addr[31:0]
  g0[3] = (unsigned)((ga >> 32) & 0x01ffffffu) | 0x80000000u;  // addr[56:32] | type=2
  v8i g1;
  g1[0] = (int)(1u << 16);                               // data_size = 2 bytes
  g1[1] = (int)((tensor_w & 0xffffu) << 16);             // tensor_dim0[15:0]
  g1[2] = (int)((tensor_w >> 16) | ((tensor_h & 0xffffu) << 16)); // dim0 hi | dim1 lo
  g1[3] = (int)((tensor_h >> 16) | (tile_w << 16));      // dim1 hi | tile_dim0
  g1[4] = (int)(tile_h & 0xffffu);                       // tile_dim1 (tile_dim2=0)
  g1[5] = (int)stride_elems;                             // tensor_dim0_stride lo32
  g1[6] = 0;                                             // stride hi | dim1_stride lo
  g1[7] = 0;
  v4i z4 = {0, 0, 0, 0};
#if __clang_major__ <= 22
  __builtin_amdgcn_tensor_load_to_lds(g0, g1, z4, z4, 0);
#else
  v8i z8 = {0, 0, 0, 0, 0, 0, 0, 0};
  __builtin_amdgcn_tensor_load_to_lds(g0, g1, z4, z4, z8, 0);
#endif
}

__device__ inline void tdm_wait() {
  __builtin_amdgcn_s_wait_tensorcnt(0);  // no-op for waves with TENSORcnt==0
}

// ---------------------------------------------------------------------------
// fp32 -> bf16 conversion (8 elems / thread)
// ---------------------------------------------------------------------------
__global__ __launch_bounds__(256) void cvt_f32_to_bf16(
    const float* __restrict__ src, bf16* __restrict__ dst, int n) {
  int i = (blockIdx.x * 256 + threadIdx.x) * 8;
  if (i + 8 > n) return;
  float4 a = *(const float4*)(src + i);
  float4 b = *(const float4*)(src + i + 4);
  v8bf o;
  o[0] = (bf16)a.x; o[1] = (bf16)a.y; o[2] = (bf16)a.z; o[3] = (bf16)a.w;
  o[4] = (bf16)b.x; o[5] = (bf16)b.y; o[6] = (bf16)b.z; o[7] = (bf16)b.w;
  *(v8bf*)(dst + i) = o;
}

// ---------------------------------------------------------------------------
// QKV GEMM: C[L, 3D] = data_bf16 @ W_qkv_bf16^T + b_qkv
// TDM streams A (128x32) and B (64x32) tiles into double-buffered LDS while
// 8 waves (4x2, 32x32 each = 2x2 WMMA tiles) run the WMMA pipe.
// Epilogue scatters to per-head Q/K/V ([head][L][128] bf16), Q scaled 1/sqrt(d)
// ---------------------------------------------------------------------------
__global__ __launch_bounds__(256) void qkv_gemm(
    const bf16* __restrict__ A, const bf16* __restrict__ W,
    const float* __restrict__ bias,
    bf16* __restrict__ Qb, bf16* __restrict__ Kb, bf16* __restrict__ Vb) {
  __shared__ bf16 As[2][BM * BK];
  __shared__ bf16 Bs[2][BN * BK];
  const int m0 = blockIdx.y * BM;
  const int n0 = blockIdx.x * BN;
  const int tid = threadIdx.x;
  const int wave = tid >> 5, lane = tid & 31;
  const int wm = wave >> 1, wn = wave & 1;
  const int lm = lane & 15, h = lane >> 4;

  const bf16* Atile = A + (size_t)m0 * DIM;
  const bf16* Btile = W + (size_t)n0 * DIM;

  v8f acc[2][2] = {};

  if (wave == 0) {
    tdm_load_2d(Atile, &As[0][0], BK, BM, DIM, DIM, L_SEQ);
    tdm_load_2d(Btile, &Bs[0][0], BK, BN, DIM, DIM, 3 * DIM);
  }
  tdm_wait();
  __syncthreads();

  const int KT = DIM / BK;
  for (int kt = 0; kt < KT; ++kt) {
    const int buf = kt & 1;
    if (wave == 0 && kt + 1 < KT) {
      tdm_load_2d(Atile + (kt + 1) * BK, &As[buf ^ 1][0], BK, BM, DIM, DIM, L_SEQ);
      tdm_load_2d(Btile + (kt + 1) * BK, &Bs[buf ^ 1][0], BK, BN, DIM, DIM, 3 * DIM);
    }
    v16bf fa0 = frag_a_ld(As[buf], BK, wm * 32,      0, lm, h);
    v16bf fa1 = frag_a_ld(As[buf], BK, wm * 32 + 16, 0, lm, h);
    v16bf fb0 = frag_b_ld(Bs[buf], BK, wn * 32,      0, lm, h);
    v16bf fb1 = frag_b_ld(Bs[buf], BK, wn * 32 + 16, 0, lm, h);
    acc[0][0] = wmma_bf16(fa0, fb0, acc[0][0]);
    acc[0][1] = wmma_bf16(fa0, fb1, acc[0][1]);
    acc[1][0] = wmma_bf16(fa1, fb0, acc[1][0]);
    acc[1][1] = wmma_bf16(fa1, fb1, acc[1][1]);
    tdm_wait();
    __syncthreads();
  }

  const float qscale = 0.08838834764831845f;  // 1/sqrt(128)
#pragma unroll
  for (int tm = 0; tm < 2; ++tm) {
#pragma unroll
    for (int tn = 0; tn < 2; ++tn) {
      const int col  = n0 + wn * 32 + tn * 16 + lm;   // [0, 6144)
      const int part = col >> 11;                      // 0=Q,1=K,2=V
      const int d2   = col & (DIM - 1);
      const int head = d2 >> 7;
      const int dd   = d2 & 127;
      bf16* dst = (part == 0) ? Qb : (part == 1 ? Kb : Vb);
      const float bv = bias[col];
      const float sc = (part == 0) ? qscale : 1.0f;
      const size_t base = (size_t)head * L_SEQ * HDIM + dd;
#pragma unroll
      for (int v = 0; v < 8; ++v) {
        const int row = m0 + wm * 32 + tm * 16 + v + 8 * h;
        dst[base + (size_t)row * HDIM] = (bf16)((acc[tm][tn][v] + bv) * sc);
      }
    }
  }
}

// ---------------------------------------------------------------------------
// Causal flash attention per head. 64 query rows / block, 128 threads (4 waves),
// each wave owns a 16-row strip. Q/K tiles streamed by TDM; V transposed in LDS
// so PV B-fragments read contiguous-K; online softmax via half-wave shuffles.
// ---------------------------------------------------------------------------
__global__ __launch_bounds__(128) void attn_kernel(
    const bf16* __restrict__ Qb, const bf16* __restrict__ Kb,
    const bf16* __restrict__ Vb, bf16* __restrict__ Ob) {
  __shared__ bf16 Qs[64 * 128];
  __shared__ bf16 Ks[64 * 128];
  __shared__ bf16 Vt[128 * 64];  // Vt[d][j]
  __shared__ bf16 Ps[64 * 64];

  const int head = blockIdx.y;
  const int ib   = blockIdx.x;
  const int i0   = ib * 64;
  const bf16* Qh = Qb + (size_t)head * L_SEQ * HDIM;
  const bf16* Kh = Kb + (size_t)head * L_SEQ * HDIM;
  const bf16* Vh = Vb + (size_t)head * L_SEQ * HDIM;

  const int tid = threadIdx.x;
  const int wave = tid >> 5, lane = tid & 31;
  const int lm = lane & 15, h = lane >> 4;
  const float NEG_INF = -__builtin_inff();

  // Q block (64x128) via TDM
  if (wave == 0)
    tdm_load_2d(Qh + (size_t)i0 * HDIM, &Qs[0], HDIM, 64, HDIM, HDIM, L_SEQ);

  v8f o[8] = {};
  float mrow[8], lrow[8];
#pragma unroll
  for (int v = 0; v < 8; ++v) { mrow[v] = NEG_INF; lrow[v] = 0.f; }

  for (int jb = 0; jb <= ib; ++jb) {
    const int j0 = jb * 64;
    __syncthreads();  // previous iteration done reading Ks/Vt
    // K block (64x128) via TDM; V block loaded + transposed cooperatively
    if (wave == 0)
      tdm_load_2d(Kh + (size_t)j0 * HDIM, &Ks[0], HDIM, 64, HDIM, HDIM, L_SEQ);
    for (int c = tid; c < 64 * 16; c += 128) {
      const int r = c >> 4, cc = (c & 15) * 8;
      v8bf vv = *(const v8bf*)(Vh + (size_t)(j0 + r) * 128 + cc);
#pragma unroll
      for (int e = 0; e < 8; ++e) Vt[(cc + e) * 64 + r] = vv[e];
    }
    tdm_wait();  // covers Q (first iter) and K
    __syncthreads();

    // S = Q K^T for this wave's 16 rows x 64 cols (scale folded into Q)
    float sv[4][8];
#pragma unroll
    for (int tn = 0; tn < 4; ++tn) {
      v8f s = {};
#pragma unroll
      for (int kk = 0; kk < 4; ++kk) {
        v16bf fa = frag_a_ld(Qs, 128, wave * 16, kk * 32, lm, h);
        v16bf fb = frag_b_ld(Ks, 128, tn * 16,   kk * 32, lm, h);
        s = wmma_bf16(fa, fb, s);
      }
#pragma unroll
      for (int v = 0; v < 8; ++v) sv[tn][v] = s[v];
    }

    // causal mask only on the diagonal block
    if (jb == ib) {
#pragma unroll
      for (int tn = 0; tn < 4; ++tn) {
        const int col = j0 + tn * 16 + lm;
#pragma unroll
        for (int v = 0; v < 8; ++v) {
          const int row = i0 + wave * 16 + v + 8 * h;
          if (col > row) sv[tn][v] = NEG_INF;
        }
      }
    }

    // online softmax (rows live across the 16 lanes of each half-wave)
    float alpha[8];
#pragma unroll
    for (int v = 0; v < 8; ++v) {
      float mx = fmaxf(fmaxf(sv[0][v], sv[1][v]), fmaxf(sv[2][v], sv[3][v]));
#pragma unroll
      for (int off = 8; off; off >>= 1) mx = fmaxf(mx, __shfl_xor(mx, off, 32));
      const float mnew = fmaxf(mrow[v], mx);
      alpha[v] = __expf(mrow[v] - mnew);
      mrow[v] = mnew;
      float rs = 0.f;
#pragma unroll
      for (int tn = 0; tn < 4; ++tn) {
        const float p = __expf(sv[tn][v] - mnew);
        sv[tn][v] = p;
        rs += p;
      }
#pragma unroll
      for (int off = 8; off; off >>= 1) rs += __shfl_xor(rs, off, 32);
      lrow[v] = lrow[v] * alpha[v] + rs;
    }

    // rescale running O
#pragma unroll
    for (int dt = 0; dt < 8; ++dt)
#pragma unroll
      for (int v = 0; v < 8; ++v) o[dt][v] *= alpha[v];

    // stage P (bf16) into this wave's private LDS strip
#pragma unroll
    for (int tn = 0; tn < 4; ++tn) {
      const int col = tn * 16 + lm;
#pragma unroll
      for (int v = 0; v < 8; ++v)
        Ps[(wave * 16 + v + 8 * h) * 64 + col] = (bf16)sv[tn][v];
    }
    __syncthreads();

    // O += P @ V
#pragma unroll
    for (int dt = 0; dt < 8; ++dt) {
#pragma unroll
      for (int kk = 0; kk < 2; ++kk) {
        v16bf fa = frag_a_ld(Ps, 64, wave * 16, kk * 32, lm, h);
        v16bf fb = frag_b_ld(Vt, 64, dt * 16,   kk * 32, lm, h);
        o[dt] = wmma_bf16(fa, fb, o[dt]);
      }
    }
  }

  // epilogue: O /= l, write bf16 into [L][D] with column = head*128 + d
  float inv[8];
#pragma unroll
  for (int v = 0; v < 8; ++v) inv[v] = 1.0f / lrow[v];
#pragma unroll
  for (int dt = 0; dt < 8; ++dt) {
    const int dd = head * HDIM + dt * 16 + lm;
#pragma unroll
    for (int v = 0; v < 8; ++v) {
      const int row = i0 + wave * 16 + v + 8 * h;
      Ob[(size_t)row * DIM + dd] = (bf16)(o[dt][v] * inv[v]);
    }
  }
}

// ---------------------------------------------------------------------------
// Output projection: out[L, D] = attn_bf16 @ W_proj_bf16^T + b_proj (fp32 out)
// Same TDM-fed WMMA structure as qkv_gemm.
// ---------------------------------------------------------------------------
__global__ __launch_bounds__(256) void proj_gemm(
    const bf16* __restrict__ A, const bf16* __restrict__ W,
    const float* __restrict__ bias, float* __restrict__ out) {
  __shared__ bf16 As[2][BM * BK];
  __shared__ bf16 Bs[2][BN * BK];
  const int m0 = blockIdx.y * BM;
  const int n0 = blockIdx.x * BN;
  const int tid = threadIdx.x;
  const int wave = tid >> 5, lane = tid & 31;
  const int wm = wave >> 1, wn = wave & 1;
  const int lm = lane & 15, h = lane >> 4;

  const bf16* Atile = A + (size_t)m0 * DIM;
  const bf16* Btile = W + (size_t)n0 * DIM;

  v8f acc[2][2] = {};

  if (wave == 0) {
    tdm_load_2d(Atile, &As[0][0], BK, BM, DIM, DIM, L_SEQ);
    tdm_load_2d(Btile, &Bs[0][0], BK, BN, DIM, DIM, DIM);
  }
  tdm_wait();
  __syncthreads();

  const int KT = DIM / BK;
  for (int kt = 0; kt < KT; ++kt) {
    const int buf = kt & 1;
    if (wave == 0 && kt + 1 < KT) {
      tdm_load_2d(Atile + (kt + 1) * BK, &As[buf ^ 1][0], BK, BM, DIM, DIM, L_SEQ);
      tdm_load_2d(Btile + (kt + 1) * BK, &Bs[buf ^ 1][0], BK, BN, DIM, DIM, DIM);
    }
    v16bf fa0 = frag_a_ld(As[buf], BK, wm * 32,      0, lm, h);
    v16bf fa1 = frag_a_ld(As[buf], BK, wm * 32 + 16, 0, lm, h);
    v16bf fb0 = frag_b_ld(Bs[buf], BK, wn * 32,      0, lm, h);
    v16bf fb1 = frag_b_ld(Bs[buf], BK, wn * 32 + 16, 0, lm, h);
    acc[0][0] = wmma_bf16(fa0, fb0, acc[0][0]);
    acc[0][1] = wmma_bf16(fa0, fb1, acc[0][1]);
    acc[1][0] = wmma_bf16(fa1, fb0, acc[1][0]);
    acc[1][1] = wmma_bf16(fa1, fb1, acc[1][1]);
    tdm_wait();
    __syncthreads();
  }

#pragma unroll
  for (int tm = 0; tm < 2; ++tm) {
#pragma unroll
    for (int tn = 0; tn < 2; ++tn) {
      const int col = n0 + wn * 32 + tn * 16 + lm;
      const float bv = bias[col];
#pragma unroll
      for (int v = 0; v < 8; ++v) {
        const int row = m0 + wm * 32 + tm * 16 + v + 8 * h;
        out[(size_t)row * DIM + col] = acc[tm][tn][v] + bv;
      }
    }
  }
}

// ---------------------------------------------------------------------------
// Launcher
// ---------------------------------------------------------------------------
extern "C" void kernel_launch(void* const* d_in, const int* in_sizes, int n_in,
                              void* d_out, int out_size, void* d_ws, size_t ws_size,
                              hipStream_t stream) {
  const float* data   = (const float*)d_in[0];
  const float* W_qkv  = (const float*)d_in[1];
  const float* b_qkv  = (const float*)d_in[2];
  const float* W_proj = (const float*)d_in[3];
  const float* b_proj = (const float*)d_in[4];
  float* out = (float*)d_out;

  char* ws = (char*)d_ws;
  const size_t MB = 1024ull * 1024ull;
  bf16* dataB  = (bf16*)(ws);            //  8 MB: data bf16
  bf16* WqkvB  = (bf16*)(ws + 8 * MB);   // 24 MB: W_qkv bf16
  bf16* WprojB = (bf16*)(ws + 32 * MB);  //  8 MB: W_proj bf16
  bf16* Qb     = (bf16*)(ws + 40 * MB);  //  8 MB: Q  [head][L][128]
  bf16* Kb     = (bf16*)(ws + 48 * MB);  //  8 MB: K
  bf16* Vb     = (bf16*)(ws + 56 * MB);  //  8 MB: V
  bf16* attnB  = (bf16*)(ws + 64 * MB);  //  8 MB: attention output bf16

  // 1. fp32 -> bf16 conversions
  cvt_f32_to_bf16<<<(L_SEQ * DIM) / (8 * 256), 256, 0, stream>>>(data, dataB, L_SEQ * DIM);
  cvt_f32_to_bf16<<<(3 * DIM * DIM) / (8 * 256), 256, 0, stream>>>(W_qkv, WqkvB, 3 * DIM * DIM);
  cvt_f32_to_bf16<<<(DIM * DIM) / (8 * 256), 256, 0, stream>>>(W_proj, WprojB, DIM * DIM);

  // 2. QKV projection (+bias, Q pre-scaled) -> per-head Q/K/V bf16
  qkv_gemm<<<dim3(3 * DIM / BN, L_SEQ / BM), 256, 0, stream>>>(
      dataB, WqkvB, b_qkv, Qb, Kb, Vb);

  // 3. causal flash attention per head
  attn_kernel<<<dim3(L_SEQ / 64, NHEADS), 128, 0, stream>>>(Qb, Kb, Vb, attnB);

  // 4. output projection (+bias) -> fp32 out
  proj_gemm<<<dim3(DIM / BN, L_SEQ / BM), 256, 0, stream>>>(
      attnB, WprojB, b_proj, out);
}
